// SubgraphEncoder_51238959841865
// MI455X (gfx1250) — compile-verified, matrix operand order checked by reading, not run on hardware
//
#include <hip/hip_runtime.h>

// ---------------- problem constants (from reference) ----------------
#define N_NODESC   50000
#define N_GRAPHSC  1000
#define NPG        50            // nodes per graph
#define EPG        600           // edges per graph
#define N_EDGESC   (N_GRAPHSC * EPG)
#define DIM        128           // H == D_IN
#define DIM2       256           // 2H
#define KEEPK      300           // ceil(0.5 * 600)

typedef __attribute__((ext_vector_type(16))) _Float16 v16h;
typedef __attribute__((ext_vector_type(8)))  _Float16 v8h;
typedef __attribute__((ext_vector_type(8)))  float    v8f;

// ---------------- weight prep: W[K][N] f32 -> Wt[N][K] f16 ----------------
__global__ void prep_weight_kernel(const float* __restrict__ W, _Float16* __restrict__ Wt,
                                   int K, int N) {
    int i = blockIdx.x * 256 + threadIdx.x;
    if (i < K * N) {
        int k = i / N, n = i % N;
        Wt[(size_t)n * K + k] = (_Float16)W[i];
    }
}

// ---------------- WMMA GEMM: Y[M,N] = X[M,K] @ W[K,N] + bias ----------------
// One wave per 16-row tile, full N in accumulators. wave32, f16 inputs, f32 accum.
// A frag (16x32 f16): lane m=lane&15, ksel=(lane>>4)*8; elems 0..7 -> K=kb+ksel..+7,
// elems 8..15 -> K=kb+16+ksel..+7.  B frag mirrors with n=lane&15 (Wt is N-major).
template <int K, int N>
__global__ __launch_bounds__(256, 1)
void gemm_wmma_bias_kernel(const float* __restrict__ X, const _Float16* __restrict__ Wt,
                           const float* __restrict__ bias, float* __restrict__ Y, int M) {
    constexpr int NT = N / 16;
    const int lane = threadIdx.x & 31;
    const int m0 = blockIdx.x * 128 + (threadIdx.x >> 5) * 16;
    if (m0 >= M) return;                       // wave-uniform: EXEC stays all-1s

    const int mrow = m0 + (lane & 15);
    const int ksel = (lane >> 4) << 3;         // 0 or 8
    const float* __restrict__ arow = X + (size_t)mrow * K;
    const int nlow = lane & 15;

    v8f acc[NT];
#pragma unroll
    for (int t = 0; t < NT; ++t)
        acc[t] = (v8f){0.f, 0.f, 0.f, 0.f, 0.f, 0.f, 0.f, 0.f};

#pragma unroll
    for (int kb = 0; kb < K; kb += 32) {
        // ---- A fragment: fp32 loads + in-register f16 convert ----
        const float* ap0 = arow + kb + ksel;
        const float* ap1 = arow + kb + 16 + ksel;
        float4 a0 = *(const float4*)(ap0);
        float4 a1 = *(const float4*)(ap0 + 4);
        float4 a2 = *(const float4*)(ap1);
        float4 a3 = *(const float4*)(ap1 + 4);
        v16h af;
        af[0]  = (_Float16)a0.x; af[1]  = (_Float16)a0.y;
        af[2]  = (_Float16)a0.z; af[3]  = (_Float16)a0.w;
        af[4]  = (_Float16)a1.x; af[5]  = (_Float16)a1.y;
        af[6]  = (_Float16)a1.z; af[7]  = (_Float16)a1.w;
        af[8]  = (_Float16)a2.x; af[9]  = (_Float16)a2.y;
        af[10] = (_Float16)a2.z; af[11] = (_Float16)a2.w;
        af[12] = (_Float16)a3.x; af[13] = (_Float16)a3.y;
        af[14] = (_Float16)a3.z; af[15] = (_Float16)a3.w;

#pragma unroll
        for (int t = 0; t < NT; ++t) {
            const _Float16* brow = Wt + (size_t)(t * 16 + nlow) * K;
            v8h blo = *(const v8h*)(brow + kb + ksel);
            v8h bhi = *(const v8h*)(brow + kb + 16 + ksel);
            v16h bf = __builtin_shufflevector(blo, bhi,
                        0, 1, 2, 3, 4, 5, 6, 7, 8, 9, 10, 11, 12, 13, 14, 15);
            acc[t] = __builtin_amdgcn_wmma_f32_16x16x32_f16(
                         false, af, false, bf, (short)0, acc[t], false, false);
        }
    }

    // ---- store D: VGPR r -> m = m0 + r + 8*(lane>>4), n = t*16 + (lane&15) ----
    const int mbase = m0 + ((lane >> 4) << 3);
#pragma unroll
    for (int t = 0; t < NT; ++t) {
        const int n = t * 16 + nlow;
        const float b = bias[n];
#pragma unroll
        for (int r = 0; r < 8; ++r)
            Y[(size_t)(mbase + r) * N + n] = acc[t][r] + b;
    }
}

// ---------------- per-row AffineNormalize(256) + ReLU, in place ----------------
__global__ void norm_relu_kernel(float* __restrict__ T, const float* __restrict__ aw,
                                 const float* __restrict__ ab, int M) {
    const int lane = threadIdx.x & 31;
    const int r = blockIdx.x * 8 + (threadIdx.x >> 5);
    if (r >= M) return;
    float* p = T + (size_t)r * DIM2;
    float4 x0 = ((float4*)p)[lane * 2];
    float4 x1 = ((float4*)p)[lane * 2 + 1];

    float s = x0.x + x0.y + x0.z + x0.w + x1.x + x1.y + x1.z + x1.w;
#pragma unroll
    for (int o = 16; o > 0; o >>= 1) s += __shfl_xor(s, o);
    const float mu = s * (1.0f / DIM2);

    float v = 0.f;
    float e;
    e = x0.x - mu; v += e * e;  e = x0.y - mu; v += e * e;
    e = x0.z - mu; v += e * e;  e = x0.w - mu; v += e * e;
    e = x1.x - mu; v += e * e;  e = x1.y - mu; v += e * e;
    e = x1.z - mu; v += e * e;  e = x1.w - mu; v += e * e;
#pragma unroll
    for (int o = 16; o > 0; o >>= 1) v += __shfl_xor(v, o);
    const float inv = rsqrtf(v * (1.0f / DIM2) + 1e-5f);

    const int n0 = lane * 8;
    float out[8] = {x0.x, x0.y, x0.z, x0.w, x1.x, x1.y, x1.z, x1.w};
#pragma unroll
    for (int j = 0; j < 8; ++j) {
        float y = aw[n0 + j] * (out[j] - mu) * inv + ab[n0 + j];
        out[j] = y > 0.f ? y : 0.f;
    }
    ((float4*)p)[lane * 2]     = (float4){out[0], out[1], out[2], out[3]};
    ((float4*)p)[lane * 2 + 1] = (float4){out[4], out[5], out[6], out[7]};
}

// ---------------- GIN aggregation: out = h + segment_sum(w * h[col], row) ------
// One block per graph. Deterministic: build in-LDS CSR sorted by (dst, edge id)
// via O(EPG^2) rank scan, then each (node,feature) sums its bucket in id order.
__global__ void gin_aggregate_kernel(const float* __restrict__ h,
                                     const int* __restrict__ rowi,
                                     const int* __restrict__ coli,
                                     const float* __restrict__ ew,   // nullable
                                     float* __restrict__ out) {
    __shared__ short dstl[EPG];
    __shared__ short srcl[EPG];
    __shared__ short order[EPG];
    __shared__ float wl[EPG];
    __shared__ int   start[NPG + 1];

    const int g = blockIdx.x, tid = threadIdx.x;
    const int nbase = g * NPG, ebase = g * EPG;

    for (int e = tid; e < EPG; e += 256) {
        dstl[e] = (short)(rowi[ebase + e] - nbase);
        srcl[e] = (short)(coli[ebase + e] - nbase);
        wl[e]   = ew ? ew[ebase + e] : 1.0f;
    }
    __syncthreads();

    for (int e = tid; e < EPG; e += 256) {
        const int d = dstl[e];
        int rank = 0;
        for (int j = 0; j < EPG; ++j) {
            const int dj = dstl[j];
            rank += (dj < d) || (dj == d && j < e);
        }
        order[rank] = (short)e;
    }
    if (tid <= NPG) {
        int s = 0;
        for (int j = 0; j < EPG; ++j) s += (dstl[j] < tid);
        start[tid] = s;
    }
    __syncthreads();

    for (int p = tid; p < NPG * DIM; p += 256) {
        const int n = p >> 7, f = p & (DIM - 1);
        float sum = h[(size_t)(nbase + n) * DIM + f];
        const int e1 = start[n + 1];
        for (int t = start[n]; t < e1; ++t) {
            const int e = order[t];
            sum += wl[e] * h[(size_t)(nbase + srcl[e]) * DIM + f];
        }
        out[(size_t)(nbase + n) * DIM + f] = sum;
    }
}

// ---------------- edge scores + global min/max (order-preserving uint) --------
__device__ __forceinline__ unsigned f2ord(float f) {
    unsigned b = __float_as_uint(f);
    return (b & 0x80000000u) ? ~b : (b | 0x80000000u);
}
__device__ __forceinline__ float ord2f(unsigned u) {
    unsigned b = (u & 0x80000000u) ? (u & 0x7FFFFFFFu) : ~u;
    return __uint_as_float(b);
}

__global__ void init_minmax_kernel(unsigned* mm) {
    if (threadIdx.x == 0) { mm[0] = 0xFFFFFFFFu; mm[1] = 0u; }
}

__global__ void edge_score_kernel(const float* __restrict__ mean,
                                  const int* __restrict__ rowi,
                                  const int* __restrict__ coli,
                                  float* __restrict__ scores, unsigned* __restrict__ mm) {
    __shared__ float smin[8], smax[8];
    const int wave = threadIdx.x >> 5, lane = threadIdx.x & 31;
    const int e = blockIdx.x * 8 + wave;

    float sc = 0.f;
    if (e < N_EDGESC) {
        const float4* a = (const float4*)(mean + (size_t)rowi[e] * DIM);
        const float4* b = (const float4*)(mean + (size_t)coli[e] * DIM);
        float4 av = a[lane], bv = b[lane];
        sc = av.x * bv.x + av.y * bv.y + av.z * bv.z + av.w * bv.w;
    }
#pragma unroll
    for (int o = 16; o > 0; o >>= 1) sc += __shfl_xor(sc, o);
    sc *= (1.0f / DIM);
    if (lane == 0) {
        if (e < N_EDGESC) { scores[e] = sc; smin[wave] = sc; smax[wave] = sc; }
        else              { smin[wave] = 3.4e38f; smax[wave] = -3.4e38f; }
    }
    __syncthreads();
    if (threadIdx.x == 0) {
        float mn = smin[0], mx = smax[0];
#pragma unroll
        for (int w = 1; w < 8; ++w) {
            mn = fminf(mn, smin[w]);
            mx = fmaxf(mx, smax[w]);
        }
        atomicMin(&mm[0], f2ord(mn));
        atomicMax(&mm[1], f2ord(mx));
    }
}

// ---------------- per-graph top-300 (stable descending), causal_w, node_mask --
__global__ void topk_kernel(const float* __restrict__ scores, const unsigned* __restrict__ mm,
                            const int* __restrict__ rowi, const int* __restrict__ coli,
                            float* __restrict__ causal_w, float* __restrict__ node_mask) {
    __shared__ float sn[EPG];
    __shared__ int flags[NPG];
    const int g = blockIdx.x, tid = threadIdx.x;
    const int nbase = g * NPG, ebase = g * EPG;

    const float mn = ord2f(mm[0]);
    const float mx = ord2f(mm[1]);
    const float denom = (mx - mn) + 1e-12f;     // same quantization as reference

    for (int e = tid; e < EPG; e += 256)
        sn[e] = (scores[ebase + e] - mn) / denom;
    if (tid < NPG) flags[tid] = 0;
    __syncthreads();

    for (int e = tid; e < EPG; e += 256) {
        const float v = sn[e];
        int cnt = 0;
        for (int j = 0; j < EPG; ++j) {
            const float u = sn[j];
            cnt += (u > v) || (u == v && j < e);   // stable descending rank
        }
        const bool keep = cnt < KEEPK;
        const int ge = ebase + e;
        causal_w[ge] = keep ? scores[ge] : 0.0f;
        if (keep) {
            flags[rowi[ge] - nbase] = 1;
            flags[coli[ge] - nbase] = 1;
        }
    }
    __syncthreads();
    if (tid < NPG) node_mask[nbase + tid] = flags[tid] ? 1.0f : 0.0f;
}

// ---------------- masked-mean readout per graph ----------------
__global__ void readout_kernel(const float* __restrict__ z, const float* __restrict__ node_mask,
                               float* __restrict__ out) {
    const int g = blockIdx.x, f = threadIdx.x;   // 128 threads
    float num = 0.f, cnt = 0.f;
#pragma unroll 5
    for (int i = 0; i < NPG; ++i) {
        const float m = node_mask[g * NPG + i];
        num += z[(size_t)(g * NPG + i) * DIM + f] * m;
        cnt += m;
    }
    out[(size_t)g * DIM + f] = num / fmaxf(cnt, 1.0f);
}

// ---------------- host-side orchestration ----------------
extern "C" void kernel_launch(void* const* d_in, const int* in_sizes, int n_in,
                              void* d_out, int out_size, void* d_ws, size_t ws_size,
                              hipStream_t stream) {
    const float* x     = (const float*)d_in[0];
    const int*   eidx  = (const int*)d_in[1];
    const int*   rowi  = eidx;
    const int*   coli  = eidx + N_EDGESC;
    const float* Wemb  = (const float*)d_in[3];
    const float* bemb  = (const float*)d_in[4];
    const float* gW1   = (const float*)d_in[5];
    const float* gb1   = (const float*)d_in[6];
    const float* gaw   = (const float*)d_in[7];
    const float* gab   = (const float*)d_in[8];
    const float* gW2   = (const float*)d_in[9];
    const float* gb2   = (const float*)d_in[10];
    const float* mW1   = (const float*)d_in[11];
    const float* mb1   = (const float*)d_in[12];
    const float* maw   = (const float*)d_in[13];
    const float* mab   = (const float*)d_in[14];
    const float* mW2   = (const float*)d_in[15];
    const float* mb2   = (const float*)d_in[16];
    const float* fW1   = (const float*)d_in[17];
    const float* fb1   = (const float*)d_in[18];
    const float* faw   = (const float*)d_in[19];
    const float* fab   = (const float*)d_in[20];
    const float* fW2   = (const float*)d_in[21];
    const float* fb2   = (const float*)d_in[22];

    // workspace layout (all 16B-aligned)
    float* A      = (float*)d_ws;                 // [50000,128] node feats / rep
    float* B      = A + (size_t)N_NODESC * DIM;   // [50000,128] aggregation buffer
    float* C      = B + (size_t)N_NODESC * DIM;   // [50000,128] mean / z ping-pong
    float* T      = C + (size_t)N_NODESC * DIM;   // [50000,256] MLP hidden
    float* scores = T + (size_t)N_NODESC * DIM2;  // [600000]
    float* causal = scores + N_EDGESC;            // [600000]
    float* nmask  = causal + N_EDGESC;            // [50000]
    unsigned* mm  = (unsigned*)(nmask + N_NODESC);// [2] (+pad)
    _Float16* wt  = (_Float16*)(mm + 8);          // [256*256] f16 transposed weights

    const int M = N_NODESC;
    const dim3 gemm_grid((M + 127) / 128);

    auto gemm = [&](const float* X, const float* W, const float* bias, float* Y,
                    int K, int N) {
        prep_weight_kernel<<<(K * N + 255) / 256, 256, 0, stream>>>(W, wt, K, N);
        if (K == 128 && N == 128)
            gemm_wmma_bias_kernel<128, 128><<<gemm_grid, 256, 0, stream>>>(X, wt, bias, Y, M);
        else if (K == 128 && N == 256)
            gemm_wmma_bias_kernel<128, 256><<<gemm_grid, 256, 0, stream>>>(X, wt, bias, Y, M);
        else
            gemm_wmma_bias_kernel<256, 128><<<gemm_grid, 256, 0, stream>>>(X, wt, bias, Y, M);
    };

    // 1) embed: A = x @ Wemb + bemb
    gemm(x, Wemb, bemb, A, DIM, DIM);

    // 2) two GIN layers (unweighted edges)
    for (int i = 0; i < 2; ++i) {
        gin_aggregate_kernel<<<N_GRAPHSC, 256, 0, stream>>>(A, rowi, coli, nullptr, B);
        gemm(B, gW1 + (size_t)i * DIM * DIM2, gb1 + (size_t)i * DIM2, T, DIM, DIM2);
        norm_relu_kernel<<<M / 8, 256, 0, stream>>>(T, gaw + (size_t)i * DIM2,
                                                    gab + (size_t)i * DIM2, M);
        gemm(T, gW2 + (size_t)i * DIM2 * DIM, gb2 + (size_t)i * DIM, A, DIM2, DIM);
    }
    // A = rep

    // 3) mean encoder: C = MLP(A)
    gemm(A, mW1, mb1, T, DIM, DIM2);
    norm_relu_kernel<<<M / 8, 256, 0, stream>>>(T, maw, mab, M);
    gemm(T, mW2, mb2, C, DIM2, DIM);

    // 4) edge scores + global min/max, then per-graph stable top-300
    init_minmax_kernel<<<1, 32, 0, stream>>>(mm);
    edge_score_kernel<<<(N_EDGESC + 7) / 8, 256, 0, stream>>>(C, rowi, coli, scores, mm);
    topk_kernel<<<N_GRAPHSC, 256, 0, stream>>>(scores, mm, rowi, coli, causal, nmask);

    // 5) three GIN feature-encoder layers with causal edge weights
    //    z0 = A(rep) -> C ; C -> A ; A -> C
    gin_aggregate_kernel<<<N_GRAPHSC, 256, 0, stream>>>(A, rowi, coli, causal, B);
    gemm(B, fW1 + 0 * (size_t)DIM * DIM2, fb1 + 0 * DIM2, T, DIM, DIM2);
    norm_relu_kernel<<<M / 8, 256, 0, stream>>>(T, faw + 0 * DIM2, fab + 0 * DIM2, M);
    gemm(T, fW2 + 0 * (size_t)DIM2 * DIM, fb2 + 0 * DIM, C, DIM2, DIM);

    gin_aggregate_kernel<<<N_GRAPHSC, 256, 0, stream>>>(C, rowi, coli, causal, B);
    gemm(B, fW1 + 1 * (size_t)DIM * DIM2, fb1 + 1 * DIM2, T, DIM, DIM2);
    norm_relu_kernel<<<M / 8, 256, 0, stream>>>(T, faw + 1 * DIM2, fab + 1 * DIM2, M);
    gemm(T, fW2 + 1 * (size_t)DIM2 * DIM, fb2 + 1 * DIM, A, DIM2, DIM);

    gin_aggregate_kernel<<<N_GRAPHSC, 256, 0, stream>>>(A, rowi, coli, causal, B);
    gemm(B, fW1 + 2 * (size_t)DIM * DIM2, fb1 + 2 * DIM2, T, DIM, DIM2);
    norm_relu_kernel<<<M / 8, 256, 0, stream>>>(T, faw + 2 * DIM2, fab + 2 * DIM2, M);
    gemm(T, fW2 + 2 * (size_t)DIM2 * DIM, fb2 + 2 * DIM, C, DIM2, DIM);

    // 6) masked-mean readout -> d_out [1000,128]
    readout_kernel<<<N_GRAPHSC, 128, 0, stream>>>(C, nmask, (float*)d_out);
}
